// Decoder_69879117906670
// MI455X (gfx1250) — compile-verified, matrix operand order checked by reading, not run on hardware
//
#include <hip/hip_runtime.h>
#include <hip/hip_bf16.h>

// ---------------- problem constants (from reference setup_inputs) -----------
#define T_LEN   128
#define BS      32
#define SRC     64
#define NTOK    32000
#define NEMB    512
#define NHID    1024
#define NATT    512
#define GATES   (4 * NHID)          // 4096
#define K0      (NEMB + NATT)       // 1024 : layer-0 input dim
#define CAT2K   (2 * NHID)          // 2048 : [h1n, ant] concat dim

#define KCH     128                 // K-chunk staged per LDS buffer (bf16 elems)
#define KPAD    (KCH + 16)          // +32B row pad: 16 tile rows land 8 banks apart

typedef __attribute__((ext_vector_type(16))) __bf16 v16bf;
typedef __attribute__((ext_vector_type(8)))  float  v8f;

// ---------------------------------------------------------------------------
// CDNA5 async-copy helpers (ASYNCcnt-tracked, no VGPR round trip).
// global_load_async_to_lds_b128: VDST = LDS byte address, VADDR = 64b global.
// ---------------------------------------------------------------------------
__device__ __forceinline__ void async_g2lds_b128(unsigned int lds_addr, const void* g) {
  asm volatile("global_load_async_to_lds_b128 %0, %1, off"
               :: "v"(lds_addr), "v"(g) : "memory");
}
__device__ __forceinline__ void wait_asynccnt0() {
  asm volatile("s_wait_asynccnt 0x0" ::: "memory");
}

// ---------------------------------------------------------------------------
// WMMA 16x16x32 bf16 operand loaders (ISA 7.12.2 16-bit 16x32 layout):
// per lane two groups of 8 contiguous bf16 (16B) -> vectorizes to *_b128.
// ---------------------------------------------------------------------------
__device__ inline v16bf load_tile_16x32(const __bf16* __restrict__ base, int lda,
                                        int r0, int k0, int lane) {
  const int half = lane >> 4;
  const int row  = r0 + (lane & 15);
  const __bf16* p = base + (size_t)row * lda + k0;
  v16bf t;
#pragma unroll
  for (int v = 0; v < 8; ++v) {
    const int kb = ((v & 4) ? 16 : 0) + 8 * half + 2 * (v & 3);
    t[2 * v]     = p[kb];
    t[2 * v + 1] = p[kb + 1];
  }
  return t;
}

template <int ROWS>
__device__ inline v16bf lds_tile_16x32(const __bf16 (&sh)[ROWS][KPAD],
                                       int r0, int kk, int lane) {
  const int half = lane >> 4;
  const int row  = r0 + (lane & 15);
  v16bf t;
#pragma unroll
  for (int v = 0; v < 8; ++v) {
    const int kb = kk + ((v & 4) ? 16 : 0) + 8 * half + 2 * (v & 3);
    t[2 * v]     = sh[row][kb];
    t[2 * v + 1] = sh[row][kb + 1];
  }
  return t;
}

// ---------------------------------------------------------------------------
// Dual-A GEMM with double-buffered async-LDS staging:
//   out[M,N] = A1[M,K1] @ W1[N,K1]^T (+ A2[M,K2] @ W2[N,K2]^T) (+ b1) (+ b2)
// Block = 256 threads = 8 waves covering a 32x64 tile; grid.x = N/64,
// grid.y = M/32.  K1, K2 multiples of KCH (all are: 512/1024/2048).
// Two sequential pipelined segments (constant base pointers at every stage
// call site -> scalar addressing, no pointer selects / spills).
// ---------------------------------------------------------------------------
__global__ void __launch_bounds__(256)
k_gemm_dual(const __bf16* __restrict__ A1, int K1, const __bf16* __restrict__ W1,
            const __bf16* __restrict__ A2, int K2, const __bf16* __restrict__ W2,
            const float* __restrict__ b1, const float* __restrict__ b2,
            float* __restrict__ out, int N) {
  const int tid  = threadIdx.x;
  const int lane = tid & 31;
  const int wave = tid >> 5;
  const int mloc = (wave & 1) * 16;
  const int nloc = (wave >> 1) * 16;
  const int mblk = blockIdx.y * 32;
  const int nblk = blockIdx.x * 64;

  __shared__ __align__(16) __bf16 shA[2][32][KPAD];   // 18 KB
  __shared__ __align__(16) __bf16 shB[2][64][KPAD];   // 36 KB

  // per-thread staging coordinates (A: 2 units of 16B; B: 4 units of 16B)
  const int arow = tid >> 4,        aug = tid & 15;        // unit 0 of A (+256 -> unit 1)
  const int n1   = K1 / KCH;
  const int n2   = A2 ? (K2 / KCH) : 0;

  auto stage = [&](int buf, const __bf16* __restrict__ As,
                   const __bf16* __restrict__ Ws, int lda, int k0) {
#pragma unroll
    for (int u = 0; u < 2; ++u) {           // A slab: 512 16B-units / 256 thr
      const int row = arow + u * 16;
      async_g2lds_b128((unsigned int)(uintptr_t)&shA[buf][row][8 * aug],
                       As + (size_t)(mblk + row) * lda + k0 + 8 * aug);
    }
#pragma unroll
    for (int u = 0; u < 4; ++u) {           // B slab: 1024 16B-units / 256 thr
      const int row = arow + u * 16;
      async_g2lds_b128((unsigned int)(uintptr_t)&shB[buf][row][8 * aug],
                       Ws + (size_t)(nblk + row) * lda + k0 + 8 * aug);
    }
  };

  auto compute = [&](int buf, v8f& acc) {
#pragma unroll
    for (int kk = 0; kk < KCH; kk += 32) {
      v16bf a = lds_tile_16x32<32>(shA[buf], mloc, kk, lane);
      v16bf b = lds_tile_16x32<64>(shB[buf], nloc, kk, lane);
      acc = __builtin_amdgcn_wmma_f32_16x16x32_bf16(false, a, false, b,
                                                    (short)0, acc, false, false);
    }
  };

  v8f acc = {};
  stage(0, A1, W1, K1, 0);
  wait_asynccnt0();
  __syncthreads();

  int ch = 0;
  for (int c = 0; c < n1; ++c, ++ch) {          // segment 1: A1/W1
    const int buf = ch & 1;
    if (c + 1 < n1)      stage(buf ^ 1, A1, W1, K1, (c + 1) * KCH);
    else if (n2 > 0)     stage(buf ^ 1, A2, W2, K2, 0);
    compute(buf, acc);
    wait_asynccnt0();    // own prefetch landed
    __syncthreads();     // everyone's landed; everyone done with old buffer
  }
  for (int c = 0; c < n2; ++c, ++ch) {          // segment 2: A2/W2
    const int buf = ch & 1;
    if (c + 1 < n2) stage(buf ^ 1, A2, W2, K2, (c + 1) * KCH);
    compute(buf, acc);
    wait_asynccnt0();
    __syncthreads();
  }

  // C/D layout: lane%16 = column, (lane>=16 ? 8 : 0)+v = row within tile
  const int n  = nblk + nloc + (lane & 15);
  const int mb = mblk + mloc + (lane >> 4) * 8;
  float bias = 0.f;
  if (b1) bias += b1[n];
  if (b2) bias += b2[n];
#pragma unroll
  for (int v = 0; v < 8; ++v)
    out[(size_t)(mb + v) * N + n] = acc[v] + bias;
}

// ---------------------------------------------------------------------------
// GEMM + tanh epilogue -> bf16 (attention projection  tanh(cat2 @ Wc^T)).
// M=32 only; direct-from-global operands (tiny K loop, L2-resident).
// ---------------------------------------------------------------------------
__global__ void k_gemm_tanh(const __bf16* __restrict__ A, int K,
                            const __bf16* __restrict__ W,
                            __bf16* __restrict__ out1, int ld1,
                            __bf16* __restrict__ out2, int ld2, int N) {
  const int lane = threadIdx.x & 31;
  const int wave = threadIdx.x >> 5;
  const int m0 = (wave & 1) * 16;
  const int n0 = blockIdx.x * 64 + (wave >> 1) * 16;

  v8f acc = {};
  for (int k = 0; k < K; k += 32) {
    v16bf a = load_tile_16x32(A, K, m0, k, lane);
    v16bf b = load_tile_16x32(W, K, n0, k, lane);
    acc = __builtin_amdgcn_wmma_f32_16x16x32_bf16(false, a, false, b,
                                                  (short)0, acc, false, false);
  }
  const int n  = n0 + (lane & 15);
  const int mb = m0 + (lane >> 4) * 8;
#pragma unroll
  for (int v = 0; v < 8; ++v) {
    float t = tanhf(acc[v]);
    __bf16 tb = (__bf16)t;
    out1[(size_t)(mb + v) * ld1 + n] = tb;
    if (out2) out2[(size_t)(mb + v) * ld2 + n] = tb;
  }
}

// ---------------------------------------------------------------------------
// Elementwise LSTM cell (gate order i,f,g,o).  gates: [BS, 4096] f32.
// ---------------------------------------------------------------------------
__global__ void k_lstm_cell(const float* __restrict__ gates,
                            const float* __restrict__ c_prev,
                            float* __restrict__ c_new,
                            float* __restrict__ h_new,
                            __bf16* __restrict__ h_bf,
                            __bf16* __restrict__ extra, int extra_ld) {
  const int idx = blockIdx.x * blockDim.x + threadIdx.x;
  if (idx >= BS * NHID) return;
  const int b = idx >> 10, j = idx & (NHID - 1);
  const float* g = gates + (size_t)b * GATES;
  const float gi = g[j], gf = g[NHID + j], gg = g[2 * NHID + j], go = g[3 * NHID + j];
  const float si = 1.f / (1.f + __expf(-gi));
  const float sf = 1.f / (1.f + __expf(-gf));
  const float so = 1.f / (1.f + __expf(-go));
  const float cn = sf * c_prev[idx] + si * tanhf(gg);
  const float hn = so * tanhf(cn);
  c_new[idx] = cn;
  h_new[idx] = hn;
  h_bf[idx]  = (__bf16)hn;
  if (extra) extra[(size_t)b * extra_ld + j] = (__bf16)hn;
}

// ---------------------------------------------------------------------------
// Fused attention per batch column: score -> mask -> softmax -> context avg.
// ---------------------------------------------------------------------------
__global__ void k_attention(const float* __restrict__ context,      // [SRC,BS,NHID]
                            const unsigned char* __restrict__ pad,  // [SRC,BS]
                            const float* __restrict__ h1,           // [BS,NHID]
                            __bf16* __restrict__ cat2) {            // [BS,CAT2K]
  const int b = blockIdx.x;
  __shared__ float sh_h[NHID];
  __shared__ float sh_s[SRC];
  for (int h = threadIdx.x; h < NHID; h += 256) sh_h[h] = h1[(size_t)b * NHID + h];
  __syncthreads();

  const int wave = threadIdx.x >> 5, lane = threadIdx.x & 31;
  for (int si = 0; si < SRC / 8; ++si) {
    const int s = wave * (SRC / 8) + si;
    const float* cp = context + ((size_t)s * BS + b) * NHID;
    float acc = 0.f;
    for (int h = lane; h < NHID; h += 32) acc += cp[h] * sh_h[h];
#pragma unroll
    for (int off = 16; off; off >>= 1) acc += __shfl_down(acc, off, 32);
    if (lane == 0)
      sh_s[s] = pad[s * BS + b] ? -__builtin_inff() : acc;
  }
  __syncthreads();
  if (threadIdx.x == 0) {
    float mx = -__builtin_inff();
    for (int s = 0; s < SRC; ++s) mx = fmaxf(mx, sh_s[s]);
    float sum = 0.f;
    for (int s = 0; s < SRC; ++s) { float e = __expf(sh_s[s] - mx); sh_s[s] = e; sum += e; }
    const float inv = 1.f / sum;
    for (int s = 0; s < SRC; ++s) sh_s[s] *= inv;
  }
  __syncthreads();
  for (int h = threadIdx.x; h < NHID; h += 256) {
    float acc = 0.f;
    for (int s = 0; s < SRC; ++s)
      acc += sh_s[s] * context[((size_t)s * BS + b) * NHID + h];
    cat2[(size_t)b * CAT2K + NHID + h] = (__bf16)acc;
  }
}

// ---------------------------------------------------------------------------
// In-place log-softmax over rows of length n (n = NTOK).
// ---------------------------------------------------------------------------
__global__ void k_logsoftmax(float* __restrict__ out, int n) {
  float* p = out + (size_t)blockIdx.x * n;
  __shared__ float red[8];
  __shared__ float s_bc;
  const int wave = threadIdx.x >> 5, lane = threadIdx.x & 31;

  float mx = -__builtin_inff();
  for (int i = threadIdx.x; i < n; i += 256) mx = fmaxf(mx, p[i]);
#pragma unroll
  for (int off = 16; off; off >>= 1) mx = fmaxf(mx, __shfl_down(mx, off, 32));
  if (lane == 0) red[wave] = mx;
  __syncthreads();
  if (threadIdx.x == 0) {
    float m = red[0];
    for (int w = 1; w < 8; ++w) m = fmaxf(m, red[w]);
    s_bc = m;
  }
  __syncthreads();
  mx = s_bc;

  float sum = 0.f;
  for (int i = threadIdx.x; i < n; i += 256) sum += __expf(p[i] - mx);
#pragma unroll
  for (int off = 16; off; off >>= 1) sum += __shfl_down(sum, off, 32);
  if (lane == 0) red[wave] = sum;
  __syncthreads();
  if (threadIdx.x == 0) {
    float s = 0.f;
    for (int w = 0; w < 8; ++w) s += red[w];
    s_bc = __logf(s);
  }
  __syncthreads();
  const float ls = s_bc;
  for (int i = threadIdx.x; i < n; i += 256) p[i] = p[i] - mx - ls;
}

// ---------------------------------------------------------------------------
__global__ void k_cvt_bf16(const float* __restrict__ src, __bf16* __restrict__ dst, int n) {
  const int i = blockIdx.x * blockDim.x + threadIdx.x;
  if (i < n) dst[i] = (__bf16)src[i];
}

__global__ void k_embed(const int* __restrict__ tgt, const float* __restrict__ emb,
                        __bf16* __restrict__ inp_all) {
  const int idx = blockIdx.x * blockDim.x + threadIdx.x;
  if (idx >= T_LEN * BS * K0) return;
  const int j = idx & (K0 - 1);
  const int tb = idx >> 10;
  float v = 0.f;
  if (j < NEMB) v = emb[(size_t)tgt[tb] * NEMB + j];
  inp_all[idx] = (__bf16)v;
}

// ---------------------------------------------------------------------------
extern "C" void kernel_launch(void* const* d_in, const int* in_sizes, int n_in,
                              void* d_out, int out_size, void* d_ws, size_t ws_size,
                              hipStream_t stream) {
  (void)in_sizes; (void)n_in; (void)out_size; (void)ws_size;
  const int*   tgt     = (const int*)  d_in[0];
  const float* h0      = (const float*)d_in[1];
  const float* c0      = (const float*)d_in[2];
  const float* context = (const float*)d_in[3];
  const unsigned char* pad = (const unsigned char*)d_in[4];
  const float* emb     = (const float*)d_in[5];
  const float* W_ih0   = (const float*)d_in[6];
  const float* W_hh0   = (const float*)d_in[7];
  const float* b_ih0   = (const float*)d_in[8];
  const float* b_hh0   = (const float*)d_in[9];
  const float* W_ih1   = (const float*)d_in[10];
  const float* W_hh1   = (const float*)d_in[11];
  const float* b_ih1   = (const float*)d_in[12];
  const float* b_hh1   = (const float*)d_in[13];
  const float* Wc      = (const float*)d_in[14];
  const float* Wd      = (const float*)d_in[15];
  const float* bd      = (const float*)d_in[16];
  float* out = (float*)d_out;

  // ---- workspace carve-up (bf16 weight mirrors + recurrent state) ----
  char* ws = (char*)d_ws;
  size_t off = 0;
  auto take = [&](size_t bytes) { char* p = ws + off; off += (bytes + 255) & ~size_t(255); return p; };
  __bf16* wih0 = (__bf16*)take((size_t)GATES * K0   * 2);
  __bf16* whh0 = (__bf16*)take((size_t)GATES * NHID * 2);
  __bf16* wih1 = (__bf16*)take((size_t)GATES * NHID * 2);
  __bf16* whh1 = (__bf16*)take((size_t)GATES * NHID * 2);
  __bf16* wc   = (__bf16*)take((size_t)NATT  * CAT2K * 2);
  __bf16* wd   = (__bf16*)take((size_t)NTOK  * NATT * 2);
  __bf16* inp_all  = (__bf16*)take((size_t)T_LEN * BS * K0 * 2);
  __bf16* att_outs = (__bf16*)take((size_t)T_LEN * BS * NATT * 2);
  float*  h_f  = (float*)take((size_t)2 * BS * NHID * 4);
  float*  c_f  = (float*)take((size_t)2 * BS * NHID * 4);
  __bf16* h_bf = (__bf16*)take((size_t)2 * BS * NHID * 2);
  float*  gts  = (float*)take((size_t)BS * GATES * 4);
  __bf16* cat2 = (__bf16*)take((size_t)BS * CAT2K * 2);

  auto cvt = [&](const float* s, __bf16* d, int n) {
    k_cvt_bf16<<<(n + 255) / 256, 256, 0, stream>>>(s, d, n);
  };
  cvt(W_ih0, wih0, GATES * K0);
  cvt(W_hh0, whh0, GATES * NHID);
  cvt(W_ih1, wih1, GATES * NHID);
  cvt(W_hh1, whh1, GATES * NHID);
  cvt(Wc,    wc,   NATT * CAT2K);
  cvt(Wd,    wd,   NTOK * NATT);

  k_embed<<<(T_LEN * BS * K0 + 255) / 256, 256, 0, stream>>>(tgt, emb, inp_all);

  hipMemcpyAsync(h_f, h0, (size_t)2 * BS * NHID * 4, hipMemcpyDeviceToDevice, stream);
  hipMemcpyAsync(c_f, c0, (size_t)2 * BS * NHID * 4, hipMemcpyDeviceToDevice, stream);
  cvt(h_f, h_bf, 2 * BS * NHID);

  const int HL = BS * NHID;
  for (int t = 0; t < T_LEN; ++t) {
    const __bf16* inp_t = inp_all + (size_t)t * BS * K0;
    k_gemm_dual<<<dim3(GATES / 64, 1), 256, 0, stream>>>(
        inp_t, K0, wih0, h_bf, NHID, whh0, b_ih0, b_hh0, gts, GATES);
    k_lstm_cell<<<(HL + 255) / 256, 256, 0, stream>>>(
        gts, c_f, c_f, h_f, h_bf, (__bf16*)nullptr, 0);
    k_gemm_dual<<<dim3(GATES / 64, 1), 256, 0, stream>>>(
        h_bf, NHID, wih1, h_bf + HL, NHID, whh1, b_ih1, b_hh1, gts, GATES);
    k_lstm_cell<<<(HL + 255) / 256, 256, 0, stream>>>(
        gts, c_f + HL, c_f + HL, h_f + HL, h_bf + HL, cat2, CAT2K);
    k_attention<<<BS, 256, 0, stream>>>(context, pad, h_f + HL, cat2);
    __bf16* feed = (t + 1 < T_LEN)
        ? inp_all + (size_t)(t + 1) * BS * K0 + NEMB : (__bf16*)nullptr;
    k_gemm_tanh<<<dim3(NATT / 64, 1), 256, 0, stream>>>(
        cat2, CAT2K, wc, att_outs + (size_t)t * BS * NATT, NATT, feed, K0, NATT);
  }

  // logits = att_outs @ Wd^T + bd, streamed straight into d_out (L2 holds Wd)
  k_gemm_dual<<<dim3(NTOK / 64, (T_LEN * BS) / 32), 256, 0, stream>>>(
      att_outs, NATT, wd, (const __bf16*)nullptr, 0, (const __bf16*)nullptr,
      bd, (const float*)nullptr, out, NTOK);
  k_logsoftmax<<<T_LEN * BS, 256, 0, stream>>>(out, NTOK);

  float* hf_dst = out + (size_t)T_LEN * BS * NTOK;
  float* cf_dst = hf_dst + (size_t)2 * BS * NHID;
  hipMemcpyAsync(hf_dst, h_f, (size_t)2 * BS * NHID * 4, hipMemcpyDeviceToDevice, stream);
  hipMemcpyAsync(cf_dst, c_f, (size_t)2 * BS * NHID * 4, hipMemcpyDeviceToDevice, stream);
}
